// CrossAttentionLayer_30683246362969
// MI455X (gfx1250) — compile-verified, matrix operand order checked by reading, not run on hardware
//
#include <hip/hip_runtime.h>

// ---------------------------------------------------------------------------
// CrossAttention for MI455X (gfx1250, wave32, WMMA bf16 + double-buffered TDM)
//   q = x1 @ Wq^T + bq ; k = x2 @ Wk^T + bk ; v = x2 @ Wv^T + bv   (bf16 ws)
//   out = softmax(q k^T / 512) v                                   (flash)
// ---------------------------------------------------------------------------

typedef __attribute__((ext_vector_type(16))) __bf16 bf16x16;
typedef __attribute__((ext_vector_type(4)))  __bf16 bf16x4;
typedef __attribute__((ext_vector_type(8)))  float  f32x8;
typedef __attribute__((ext_vector_type(4)))  float  f32x4;
typedef __attribute__((ext_vector_type(4)))  unsigned int u32x4;
typedef __attribute__((ext_vector_type(4)))  int    i32x4;
typedef __attribute__((ext_vector_type(8)))  int    i32x8;

#define D_DIM 512
#define MTOT  (8 * 2048)

#if defined(__has_builtin)
#  if __has_builtin(__builtin_amdgcn_tensor_load_to_lds) && \
      __has_builtin(__builtin_amdgcn_s_wait_tensorcnt)
#    define HAVE_TDM 1
#  endif
#endif
#ifndef HAVE_TDM
#  define HAVE_TDM 0
#endif

// Compiler-only scheduling fence: keep global loads issued before converts.
#define SCHED_FENCE() asm volatile("" ::: "memory")

// 16x32 bf16 A/B fragment: lanes 0-15 hold K[0..7] and K[16..23] of row/col
// (lane&15); lanes 16-31 hold K[8..15] and K[24..31].  Two 16B LDS loads.
__device__ __forceinline__ bf16x16 load_frag(const __bf16* lo, const __bf16* hi) {
  union { bf16x16 v; f32x4 f[2]; } u;
  u.f[0] = *(const f32x4*)lo;
  u.f[1] = *(const f32x4*)hi;
  return u.v;
}

__device__ __forceinline__ f32x8 wmma_bf16(bf16x16 a, bf16x16 b, f32x8 c) {
  return __builtin_amdgcn_wmma_f32_16x16x32_bf16(false, a, false, b,
                                                 (short)0, c, false, false);
}

#if HAVE_TDM
// 2D TDM tile load: global (row-major, stride_e elements, 2B elements) -> LDS
// with per-row padding (pad_int_code 4 => pad after 32 DWORDs = 128B row;
// pad_amt_code 3 => 4 DWORDs = 16B pad  => 72-element LDS row stride).
__device__ __forceinline__ void tdm_load_2d(const void* gaddr, unsigned lds_off,
                                            unsigned tile_w, unsigned tile_h,
                                            unsigned stride_e,
                                            unsigned pad_int_code,
                                            unsigned pad_amt_code) {
  unsigned long long ga = (unsigned long long)gaddr;
  u32x4 g0;
  g0[0] = 1u;                                              // count=1, user D#
  g0[1] = lds_off;                                         // LDS byte address
  g0[2] = (unsigned)ga;                                    // global addr lo
  g0[3] = (unsigned)((ga >> 32) & 0x1FFFFFFu) | (2u << 30);// addr hi | type=2
  i32x8 g1;
  g1[0] = (int)((1u << 16) |                               // data_size = 2B
                (1u << 20) |                               // pad_enable
                (pad_int_code << 22) | (pad_amt_code << 25));
  g1[1] = (int)((stride_e & 0xFFFFu) << 16);               // tensor_dim0 lo16
  g1[2] = (int)(((stride_e >> 16) & 0xFFFFu) |             // tensor_dim0 hi16
                (0x4000u << 16));                          // tensor_dim1 lo16
  g1[3] = (int)((tile_w & 0xFFFFu) << 16);                 // dim1 hi=0, tile_dim0
  g1[4] = (int)(tile_h & 0xFFFFu);                         // tile_dim1, dim2=0
  g1[5] = (int)stride_e;                                   // tensor_dim0_stride
  g1[6] = 0;
  g1[7] = 0;
  i32x4 z4 = {0, 0, 0, 0};
#if __clang_major__ >= 23
  i32x8 z8 = {0, 0, 0, 0, 0, 0, 0, 0};
  __builtin_amdgcn_tensor_load_to_lds(g0, g1, z4, z4, z8, 0);
#else
  __builtin_amdgcn_tensor_load_to_lds(g0, g1, z4, z4, 0);
#endif
}
#endif

// ---------------------------------------------------------------------------
// Projection: Y[m,e] = sum_d X[m,d]*W[e,d] + bias[e], Y in bf16.
// 64x64 block tile, K staged in 64-chunks, 8 waves = 4(m) x 2(n-half).
// ---------------------------------------------------------------------------
__global__ __launch_bounds__(256)
void proj_bf16_kernel(const float* __restrict__ X, const float* __restrict__ W,
                      const float* __restrict__ bias, __bf16* __restrict__ Y) {
  __shared__ alignas(16) __bf16 a_lds[64 * 72];
  __shared__ alignas(16) __bf16 b_lds[64 * 72];

  const int tid  = threadIdx.x;
  const int lane = tid & 31;
  const int w    = tid >> 5;
  const int hi8  = (lane >> 4) << 3;
  const int l16  = lane & 15;
  const int m0   = blockIdx.x * 64;
  const int n0   = blockIdx.y * 64;
  const int wm   = w & 3;
  const int wnh  = w >> 2;

  f32x8 c0 = {}; f32x8 c1 = {};

  for (int kc = 0; kc < 8; ++kc) {
    // ---- batch global loads (8 in flight), then convert + store -----------
    f32x4 fa[4], fb[4];
    #pragma unroll
    for (int j = 0; j < 4; ++j) {
      int vi  = tid + j * 256;
      int row = vi >> 4, c4 = vi & 15;
      fa[j] = *(const f32x4*)&X[(size_t)(m0 + row) * D_DIM + kc * 64 + c4 * 4];
      fb[j] = *(const f32x4*)&W[(size_t)(n0 + row) * D_DIM + kc * 64 + c4 * 4];
    }
    SCHED_FENCE();
    #pragma unroll
    for (int j = 0; j < 4; ++j) {
      int vi  = tid + j * 256;
      int row = vi >> 4, c4 = vi & 15;
      bf16x4 ha; ha[0]=(__bf16)fa[j].x; ha[1]=(__bf16)fa[j].y; ha[2]=(__bf16)fa[j].z; ha[3]=(__bf16)fa[j].w;
      bf16x4 hb; hb[0]=(__bf16)fb[j].x; hb[1]=(__bf16)fb[j].y; hb[2]=(__bf16)fb[j].z; hb[3]=(__bf16)fb[j].w;
      *(bf16x4*)&a_lds[row * 72 + c4 * 4] = ha;
      *(bf16x4*)&b_lds[row * 72 + c4 * 4] = hb;
    }
    __syncthreads();
    #pragma unroll
    for (int ks = 0; ks < 2; ++ks) {
      const __bf16* ar  = &a_lds[(wm * 16 + l16) * 72 + ks * 32];
      const __bf16* br0 = &b_lds[(wnh * 32 + l16) * 72 + ks * 32];
      const __bf16* br1 = &b_lds[(wnh * 32 + 16 + l16) * 72 + ks * 32];
      bf16x16 af  = load_frag(ar + hi8, ar + 16 + hi8);
      bf16x16 bf0 = load_frag(br0 + hi8, br0 + 16 + hi8);
      bf16x16 bf1 = load_frag(br1 + hi8, br1 + 16 + hi8);
      c0 = wmma_bf16(af, bf0, c0);
      c1 = wmma_bf16(af, bf1, c1);
    }
    __syncthreads();
  }

  const int col0 = n0 + wnh * 32 + l16;
  const int col1 = col0 + 16;
  const float bv0 = bias[col0];
  const float bv1 = bias[col1];
  #pragma unroll
  for (int r = 0; r < 8; ++r) {
    size_t grow = (size_t)(m0 + wm * 16 + hi8 + r);
    Y[grow * D_DIM + col0] = (__bf16)(c0[r] + bv0);
    Y[grow * D_DIM + col1] = (__bf16)(c1[r] + bv1);
  }
}

// ---------------------------------------------------------------------------
// Flash attention: 64 q-rows per block, stream 32-key tiles over S2=2048.
// LDS (54 KB):
//   region0 (union):  [q0 9216 | k0 4608 | q1 9216 | k1 4608]  /  [v^T 40960]
//   s 64x33 f32, p 64x40 bf16, m/l/alpha 64 f32 each
// Q/K chunks are double-buffered TDM DMAs (issued by wave 0): chunk dc+1
// flies while all 8 waves run WMMA on chunk dc.
// ---------------------------------------------------------------------------
__global__ __launch_bounds__(256)
void attn_kernel(const __bf16* __restrict__ Q, const __bf16* __restrict__ K,
                 const __bf16* __restrict__ V, float* __restrict__ O) {
  __shared__ alignas(16) unsigned char smem[55296];
  __bf16* qb[2] = { (__bf16*)smem,           (__bf16*)(smem + 13824) };
  __bf16* kb[2] = { (__bf16*)(smem + 9216),  (__bf16*)(smem + 23040) };
  __bf16* vt  = (__bf16*)smem;                 // 512*40*2 = 40960 (V-phase union)
  float*  sS  = (float*)(smem + 40960);        // 64*33*4  = 8448
  __bf16* pP  = (__bf16*)(smem + 49408);       // 64*40*2  = 5120
  float*  mI  = (float*)(smem + 54528);
  float*  lI  = (float*)(smem + 54784);
  float*  aI  = (float*)(smem + 55040);

  const int tid  = threadIdx.x;
  const int lane = tid & 31;
  const int w    = tid >> 5;
  const int hi8  = (lane >> 4) << 3;
  const int l16  = lane & 15;
  const int wm   = w & 3;
  const int wnh  = w >> 2;

  const int    bidx  = blockIdx.x;
  const size_t qrow0 = (size_t)(bidx >> 5) * 2048 + (size_t)(bidx & 31) * 64;
  const size_t krow0 = (size_t)(bidx >> 5) * 2048;

#if HAVE_TDM
  const unsigned qoff[2] = { (unsigned)(unsigned long long)qb[0],
                             (unsigned)(unsigned long long)qb[1] };
  const unsigned koff[2] = { (unsigned)(unsigned long long)kb[0],
                             (unsigned)(unsigned long long)kb[1] };
#endif

  if (tid < 64) { mI[tid] = -1e30f; lI[tid] = 0.0f; }

  f32x8 acc[16];
  #pragma unroll
  for (int t = 0; t < 16; ++t) acc[t] = (f32x8){};
  __syncthreads();

  for (int kt = 0; kt < 64; ++kt) {
    const size_t kr0 = krow0 + (size_t)kt * 32;

    // L2 prefetch of next key/value tiles (global_prefetch_b8)
    if (kt + 1 < 64) {
      __builtin_prefetch(&K[(kr0 + 32) * D_DIM] + (size_t)tid * 64, 0, 1);
      __builtin_prefetch(&V[(kr0 + 32) * D_DIM] + (size_t)tid * 64, 0, 1);
    }

    // ---- S = q k^T over d=512, double-buffered 64-wide d chunks -----------
#if HAVE_TDM
    if (tid == 0) {                 // prologue DMA: chunk 0 into buffer 0
      tdm_load_2d(&Q[qrow0 * D_DIM], qoff[0], 64, 64, D_DIM, 4, 3);
      tdm_load_2d(&K[kr0   * D_DIM], koff[0], 64, 32, D_DIM, 4, 3);
    }
#endif
    f32x8 cs = {};
    for (int dc = 0; dc < 8; ++dc) {
      const int cur = dc & 1;
#if HAVE_TDM
      if (tid == 0) __builtin_amdgcn_s_wait_tensorcnt(0);  // buffer `cur` ready
      __syncthreads();
      if (tid == 0 && dc < 7) {     // kick chunk dc+1 while we compute dc
        tdm_load_2d(&Q[qrow0 * D_DIM + (dc + 1) * 64], qoff[cur ^ 1], 64, 64, D_DIM, 4, 3);
        tdm_load_2d(&K[kr0   * D_DIM + (dc + 1) * 64], koff[cur ^ 1], 64, 32, D_DIM, 4, 3);
      }
#else
      #pragma unroll
      for (int j = 0; j < 2; ++j) {            // q chunk 64x64
        int vi = tid + j * 256;
        int row = vi >> 3, c8 = vi & 7;
        *(f32x4*)&qb[cur][row * 72 + c8 * 8] =
            *(const f32x4*)&Q[(qrow0 + row) * D_DIM + dc * 64 + c8 * 8];
      }
      {                                        // k chunk 32x64
        int row = tid >> 3, c8 = tid & 7;
        *(f32x4*)&kb[cur][row * 72 + c8 * 8] =
            *(const f32x4*)&K[(kr0 + row) * D_DIM + dc * 64 + c8 * 8];
      }
      __syncthreads();
#endif
      #pragma unroll
      for (int ks = 0; ks < 2; ++ks) {
        const __bf16* qr = &qb[cur][(wm * 16 + l16) * 72 + ks * 32];
        const __bf16* kr = &kb[cur][(wnh * 16 + l16) * 72 + ks * 32];
        bf16x16 aq  = load_frag(qr + hi8, qr + 16 + hi8);
        bf16x16 bkf = load_frag(kr + hi8, kr + 16 + hi8);
        cs = wmma_bf16(aq, bkf, cs);
      }
#if !HAVE_TDM
      __syncthreads();              // TDM path: next-iter top barrier suffices
#endif
    }
    {                                          // scores -> LDS, scaled by 1/D
      int scol = wnh * 16 + l16;
      #pragma unroll
      for (int r = 0; r < 8; ++r)
        sS[(wm * 16 + hi8 + r) * 33 + scol] = cs[r] * (1.0f / 512.0f);
    }
    __syncthreads();

    // ---- online softmax: one thread per q row -----------------------------
    if (tid < 64) {
      float mo = mI[tid], lo = lI[tid];
      float mx = mo;
      const float* srow = &sS[tid * 33];
      #pragma unroll
      for (int j2 = 0; j2 < 32; ++j2) mx = fmaxf(mx, srow[j2]);
      float alpha = __expf(mo - mx);
      float sum = 0.0f;
      __bf16* prow = &pP[tid * 40];
      #pragma unroll
      for (int j2 = 0; j2 < 32; ++j2) {
        float pv = __expf(srow[j2] - mx);
        prow[j2] = (__bf16)pv;
        sum += pv;
      }
      mI[tid] = mx; lI[tid] = lo * alpha + sum; aI[tid] = alpha;
    }

    // ---- stage V^T: batch 8 loads in flight, then transpose-scatter -------
    {
      f32x4 vreg[8];
      #pragma unroll
      for (int j = 0; j < 8; ++j) {
        int vi = tid + j * 256;
        int row = vi >> 6, c8 = vi & 63;
        vreg[j] = *(const f32x4*)&V[(kr0 + row) * D_DIM + c8 * 8];
      }
      SCHED_FENCE();
      #pragma unroll
      for (int j = 0; j < 8; ++j) {
        int vi = tid + j * 256;
        int row = vi >> 6, c8 = vi & 63;
        union { f32x4 f; __bf16 h[8]; } u;
        u.f = vreg[j];
        #pragma unroll
        for (int e = 0; e < 8; ++e) vt[(c8 * 8 + e) * 40 + row] = u.h[e];
      }
    }
    __syncthreads();

    // ---- rescale running output, then acc += P V --------------------------
    {
      int rb = wm * 16 + hi8;
      f32x4 a0 = *(const f32x4*)&aI[rb];
      f32x4 a1 = *(const f32x4*)&aI[rb + 4];
      #pragma unroll
      for (int t = 0; t < 16; ++t) {
        acc[t][0] *= a0.x; acc[t][1] *= a0.y; acc[t][2] *= a0.z; acc[t][3] *= a0.w;
        acc[t][4] *= a1.x; acc[t][5] *= a1.y; acc[t][6] *= a1.z; acc[t][7] *= a1.w;
      }
    }
    {
      const __bf16* pr = &pP[(wm * 16 + l16) * 40];
      bf16x16 pa = load_frag(pr + hi8, pr + 16 + hi8);
      #pragma unroll
      for (int t = 0; t < 16; ++t) {
        const __bf16* vr = &vt[(wnh * 256 + t * 16 + l16) * 40];
        bf16x16 vb = load_frag(vr + hi8, vr + 16 + hi8);
        acc[t] = wmma_bf16(pa, vb, acc[t]);
      }
    }
    __syncthreads();   // protect vt/p (and TDM target region) for next tile
  }

  // ---- epilogue: divide by row sums, store fp32 ---------------------------
  {
    int rb = wm * 16 + hi8;
    f32x4 l0 = *(const f32x4*)&lI[rb];
    f32x4 l1 = *(const f32x4*)&lI[rb + 4];
    float inv[8] = {1.0f / l0.x, 1.0f / l0.y, 1.0f / l0.z, 1.0f / l0.w,
                    1.0f / l1.x, 1.0f / l1.y, 1.0f / l1.z, 1.0f / l1.w};
    #pragma unroll
    for (int t = 0; t < 16; ++t) {
      int col = wnh * 256 + t * 16 + l16;
      #pragma unroll
      for (int r = 0; r < 8; ++r)
        O[(qrow0 + wm * 16 + hi8 + r) * D_DIM + col] = acc[t][r] * inv[r];
    }
  }
}

// ---------------------------------------------------------------------------
extern "C" void kernel_launch(void* const* d_in, const int* in_sizes, int n_in,
                              void* d_out, int out_size, void* d_ws, size_t ws_size,
                              hipStream_t stream) {
  (void)in_sizes; (void)n_in; (void)out_size; (void)ws_size;
  const float* x1 = (const float*)d_in[0];
  const float* x2 = (const float*)d_in[1];
  const float* Wq = (const float*)d_in[2];
  const float* bq = (const float*)d_in[3];
  const float* Wk = (const float*)d_in[4];
  const float* bk = (const float*)d_in[5];
  const float* Wv = (const float*)d_in[6];
  const float* bv = (const float*)d_in[7];
  float* out = (float*)d_out;

  // bf16 workspace: q | k | v, each [16384, 512] -> 16 MB, 48 MB total
  __bf16* qws = (__bf16*)d_ws;
  __bf16* kws = qws + (size_t)MTOT * D_DIM;
  __bf16* vws = kws + (size_t)MTOT * D_DIM;

  dim3 pgrid(MTOT / 64, D_DIM / 64);
  proj_bf16_kernel<<<pgrid, 256, 0, stream>>>(x1, Wq, bq, qws);
  proj_bf16_kernel<<<pgrid, 256, 0, stream>>>(x2, Wk, bk, kws);
  proj_bf16_kernel<<<pgrid, 256, 0, stream>>>(x2, Wv, bv, vws);

  attn_kernel<<<MTOT / 64, 256, 0, stream>>>(qws, kws, vws, out);
}
